// CGCNNConv_63462436766119
// MI455X (gfx1250) — matrix operand order repaired and checked by reading, not compiled, not run on hardware
//
#include <hip/hip_runtime.h>
#include <cstdint>

#define N_ATOMS   50000
#define N_EDGES   800000
#define NODE_DIM  64
#define EDGE_DIM  64
#define KDIM      192
#define NOUT      128
#define NTILES    (N_EDGES / 16)
#define BN_EPS    1e-5f
#define LDSROW    200      // padded z-row stride in bf16 elements (192 + 8 pad)

typedef __attribute__((ext_vector_type(16))) __bf16 v16bf;
typedef __attribute__((ext_vector_type(8)))  __bf16 v8bf;
typedef __attribute__((ext_vector_type(4)))  __bf16 v4bf;
typedef __attribute__((ext_vector_type(8)))  float  v8f;
typedef __attribute__((ext_vector_type(4)))  float  v4f;

// bf16 hi/lo split: x ~= (float)hi + (float)lo; accumulating
// hi*hi + hi*lo + lo*hi on the bf16 WMMA pipe gives ~fp32 accuracy.
__device__ __forceinline__ __bf16 bf_hi(float x) { return (__bf16)x; }
__device__ __forceinline__ __bf16 bf_lo(float x) {
    return (__bf16)(x - (float)((__bf16)x));
}

// ---------------------------------------------------------------------------
// Kernel 1: zero the BN stats accumulators (sum[128], sumsq[128])
// ---------------------------------------------------------------------------
__global__ void cgcnn_zero_stats(float* __restrict__ stats) {
    stats[threadIdx.x] = 0.0f;   // 256 threads, 256 floats
}

// ---------------------------------------------------------------------------
// Kernel 2: fused gather + dual-GEMM (bf16x3 WMMA) + BN partial stats.
// Block = 128 threads = 4 waves. Wave w owns output columns [32w, 32w+32)
// (cols 0..63 = core pre-activation, 64..127 = filter pre-activation).
// z tile is split into bf16 hi/lo planes ONCE at staging time; the compute
// loop reads A fragments as packed bf16 via 16B LDS loads (no per-wave
// conversion VALU on the critical path).
// ---------------------------------------------------------------------------
__global__ __launch_bounds__(128) void cgcnn_gemm_stats(
    const float* __restrict__ atom,    // [N_ATOMS, 64]
    const float* __restrict__ edgef,   // [N_EDGES, 64]
    const float* __restrict__ Wcore,   // [64, 192]
    const float* __restrict__ Wfilt,   // [64, 192]
    const int*   __restrict__ eidx,    // [N_EDGES, 2] (src, dst)
    float*       __restrict__ Y,       // [N_EDGES, 128] pre-activations
    float*       __restrict__ stats)   // [256] col sums, col sumsq
{
    __shared__ __align__(16) __bf16 zhi[16 * LDSROW];
    __shared__ __align__(16) __bf16 zlo[16 * LDSROW];

    const int tid  = threadIdx.x;
    const int lane = tid & 31;
    const int wave = tid >> 5;
    const int lrow = lane & 15;      // row (A/C) or column (B) selector
    const int lhi  = lane >> 4;      // upper-half-of-wave flag

    // ---- Preload B fragments (hi/lo bf16) for this wave's 32 columns ----
    // One-time cost, amortized over ~40 tiles per block.
    // B k order used here (contiguous, same as A): lane n<16 holds W[col=n][klo..klo+15],
    // lanes 16..31 hold klo+16..klo+31 -> 16 contiguous floats per lane.
    v16bf Bhi[2][6], Blo[2][6];
    #pragma unroll
    for (int c = 0; c < 2; ++c) {
        const int n = wave * 32 + c * 16 + lrow;          // combined col 0..127
        const float* wrow = (n < 64) ? (Wcore + n * KDIM)
                                     : (Wfilt + (n - 64) * KDIM);
        #pragma unroll
        for (int kb = 0; kb < 6; ++kb) {
            const int klo = kb * 32 + lhi * 16;
            #pragma unroll
            for (int i = 0; i < 16; ++i) {
                const float x = wrow[klo + i];
                Bhi[c][kb][i] = bf_hi(x);
                Blo[c][kb][i] = bf_lo(x);
            }
        }
    }

    float s[2]  = {0.0f, 0.0f};      // per-lane column partial sums
    float s2[2] = {0.0f, 0.0f};      // per-lane column partial sum of squares

    for (int t = blockIdx.x; t < NTILES; t += (int)gridDim.x) {
        // ---- stage z tile: gather 16 edges x 192 f32, split to bf16 hi/lo ----
        #pragma unroll
        for (int i = 0; i < 6; ++i) {
            const int idx = tid + i * 128;      // 0..767 float4 slots
            const int row = idx / 48;
            const int j   = idx - row * 48;     // float4 index within row
            const int e   = t * 16 + row;
            const v4f* src4;
            if (j < 16) {                       // segment 0: atom[dst]
                const int d = eidx[e * 2 + 1];
                src4 = (const v4f*)(atom + (size_t)d * NODE_DIM) + j;
            } else if (j < 32) {                // segment 1: atom[src]
                const int sr = eidx[e * 2 + 0];
                src4 = (const v4f*)(atom + (size_t)sr * NODE_DIM) + (j - 16);
            } else {                            // segment 2: edge features
                src4 = (const v4f*)(edgef + (size_t)e * EDGE_DIM) + (j - 32);
            }
            const v4f x = *src4;
            v4bf h, l;
            #pragma unroll
            for (int q = 0; q < 4; ++q) {
                const __bf16 hq = bf_hi(x[q]);
                h[q] = hq;
                l[q] = (__bf16)(x[q] - (float)hq);
            }
            *(v4bf*)(&zhi[row * LDSROW + j * 4]) = h;
            *(v4bf*)(&zlo[row * LDSROW + j * 4]) = l;
        }

        // ---- prefetch next tile's HBM stream (overlaps WMMA phase) ----
        const int tn = t + (int)gridDim.x;
        if (tn < NTILES) {
            if (tid < 16)
                __builtin_prefetch(edgef + (size_t)(tn * 16 + tid) * EDGE_DIM, 0, 0);
            if (tid == 16)
                __builtin_prefetch(eidx + (size_t)tn * 32, 0, 0);
        }
        __syncthreads();

        // ---- 16x32x128 tile GEMM over K=192 (6 k-blocks, bf16x3) ----
        v8f acc[2];
        #pragma unroll
        for (int c = 0; c < 2; ++c)
            #pragma unroll
            for (int v = 0; v < 8; ++v) acc[c][v] = 0.0f;

        #pragma unroll
        for (int kb = 0; kb < 6; ++kb) {
            // A k order used here (contiguous, same as B): lane holds K run {k0..k0+15},
            // k0 = kb*32 + 16*lhi -> two 16B LDS loads/plane.
            const int base = lrow * LDSROW + kb * 32 + lhi * 16;
            const v8bf h0 = *(const v8bf*)(&zhi[base]);
            const v8bf h1 = *(const v8bf*)(&zhi[base + 8]);
            const v8bf l0 = *(const v8bf*)(&zlo[base]);
            const v8bf l1 = *(const v8bf*)(&zlo[base + 8]);
            const v16bf ahi = __builtin_shufflevector(
                h0, h1, 0,1,2,3,4,5,6,7,8,9,10,11,12,13,14,15);
            const v16bf alo = __builtin_shufflevector(
                l0, l1, 0,1,2,3,4,5,6,7,8,9,10,11,12,13,14,15);
            #pragma unroll
            for (int c = 0; c < 2; ++c) {
                acc[c] = __builtin_amdgcn_wmma_f32_16x16x32_bf16(
                    false, ahi, false, Bhi[c][kb], (short)0, acc[c], false, false);
                acc[c] = __builtin_amdgcn_wmma_f32_16x16x32_bf16(
                    false, ahi, false, Blo[c][kb], (short)0, acc[c], false, false);
                acc[c] = __builtin_amdgcn_wmma_f32_16x16x32_bf16(
                    false, alo, false, Bhi[c][kb], (short)0, acc[c], false, false);
            }
        }
        __syncthreads();   // LDS reads done; next iteration may restage

        // ---- write Y tile + accumulate BN stats ----
        // C/D layout: VGPR v holds M = v + 8*lhi, N = lrow.
        #pragma unroll
        for (int c = 0; c < 2; ++c) {
            const int col = wave * 32 + c * 16 + lrow;
            #pragma unroll
            for (int v = 0; v < 8; ++v) {
                const int row = lhi * 8 + v;
                const float val = acc[c][v];
                Y[(size_t)(t * 16 + row) * NOUT + col] = val;
                s[c]  += val;
                s2[c] += val * val;
            }
        }
    }

    // ---- flush per-lane stats (each lane's column is fixed) ----
    #pragma unroll
    for (int c = 0; c < 2; ++c) {
        const int col = wave * 32 + c * 16 + lrow;
        atomicAdd(&stats[col],       s[c]);
        atomicAdd(&stats[128 + col], s2[c]);
    }
}

// ---------------------------------------------------------------------------
// Kernel 3: finalize BN -> per-column scale/shift.
// Bias terms cancel inside BatchNorm, so they never appear anywhere.
// ---------------------------------------------------------------------------
__global__ void cgcnn_bn_finalize(
    const float* __restrict__ gfil, const float* __restrict__ bfil,
    const float* __restrict__ gcore, const float* __restrict__ bcore,
    float* __restrict__ stats)       // [0:128) sum, [128:256) sumsq,
                                     // out: [256:384) scale, [384:512) shift
{
    const int col = threadIdx.x;     // 0..127
    const float inv_n = 1.0f / (float)N_EDGES;
    const float mean = stats[col] * inv_n;
    const float var  = stats[128 + col] * inv_n - mean * mean;
    float g, b;
    if (col < 64) { g = gcore[col];      b = bcore[col]; }
    else          { g = gfil[col - 64];  b = bfil[col - 64]; }
    const float sc = g * rsqrtf(var + BN_EPS);
    stats[256 + col] = sc;
    stats[384 + col] = b - mean * sc;
}

// ---------------------------------------------------------------------------
// Kernel 4: out = atom_features (segment-sum identity term)
// ---------------------------------------------------------------------------
__global__ void cgcnn_init_out(const float* __restrict__ atom,
                               float* __restrict__ out) {
    const int i = blockIdx.x * blockDim.x + threadIdx.x;   // float4 index
    ((v4f*)out)[i] = ((const v4f*)atom)[i];
}

// ---------------------------------------------------------------------------
// Kernel 5: activations + atomic scatter-add into dst atoms.
// One thread per (edge, column-pair): 2 columns each.
// ---------------------------------------------------------------------------
__global__ __launch_bounds__(256) void cgcnn_epilogue(
    const float* __restrict__ Y, const float* __restrict__ stats,
    const int* __restrict__ eidx, float* __restrict__ out)
{
    const int idx = blockIdx.x * 256 + threadIdx.x;
    const int e  = idx >> 5;
    const int cp = idx & 31;
    if (e >= N_EDGES) return;
    const int dst = eidx[e * 2 + 1];
    const float* yrow = Y + (size_t)e * NOUT;
    float* orow = out + (size_t)dst * NODE_DIM;
    #pragma unroll
    for (int u = 0; u < 2; ++u) {
        const int col = cp * 2 + u;
        const float xs = stats[256 + col] * yrow[col]      + stats[384 + col];
        const float xg = stats[320 + col] * yrow[64 + col] + stats[448 + col];
        const float sig  = (xs > 20.0f) ? xs : log1pf(__expf(xs)); // softplus
        const float gate = 1.0f / (1.0f + __expf(-xg));            // sigmoid
        atomicAdd(orow + col, gate * sig);
    }
}

// ---------------------------------------------------------------------------
extern "C" void kernel_launch(void* const* d_in, const int* in_sizes, int n_in,
                              void* d_out, int out_size, void* d_ws, size_t ws_size,
                              hipStream_t stream)
{
    (void)in_sizes; (void)n_in; (void)out_size; (void)ws_size;
    const float* atom  = (const float*)d_in[0];
    const float* edgef = (const float*)d_in[1];
    const float* Wfilt = (const float*)d_in[2];
    // d_in[3] = b_filter : cancels in BatchNorm, unused
    const float* gfil  = (const float*)d_in[4];
    const float* bfil  = (const float*)d_in[5];
    const float* Wcore = (const float*)d_in[6];
    // d_in[7] = b_core : cancels in BatchNorm, unused
    const float* gcore = (const float*)d_in[8];
    const float* bcore = (const float*)d_in[9];
    const int*   eidx  = (const int*)d_in[10];

    float* out   = (float*)d_out;
    float* stats = (float*)d_ws;          // 512 floats
    float* Y     = stats + 512;           // [N_EDGES, 128] pre-activations

    cgcnn_zero_stats<<<1, 256, 0, stream>>>(stats);
    cgcnn_gemm_stats<<<1250, 128, 0, stream>>>(atom, edgef, Wcore, Wfilt,
                                               eidx, Y, stats);
    cgcnn_bn_finalize<<<1, 128, 0, stream>>>(gfil, bfil, gcore, bcore, stats);
    cgcnn_init_out<<<(N_ATOMS * NODE_DIM / 4) / 256, 256, 0, stream>>>(atom, out);
    cgcnn_epilogue<<<(N_EDGES * 32) / 256, 256, 0, stream>>>(Y, stats, eidx, out);
}
